// CrossAttention_9869834846452
// MI455X (gfx1250) — matrix operand order repaired and checked by reading, not compiled, not run on hardware
//
#include <hip/hip_runtime.h>

#define BB 4
#define NN 4096
#define CC 768
#define HH 12
#define PP 1024
#define HD 64
#define QK_SCALE 0.125f   // 64^-0.5
#define ATT_WAVES 8

typedef __attribute__((ext_vector_type(16))) __bf16 bf16x16;
typedef __attribute__((ext_vector_type(8)))  __bf16 bf16x8;
typedef __attribute__((ext_vector_type(4)))  __bf16 bf16x4;
typedef __attribute__((ext_vector_type(8)))  float  f32x8;
typedef __attribute__((ext_vector_type(4)))  float  f32x4;

union BF16x16U { bf16x16 v; bf16x8 h[2]; };

#define WMMA_BF16(A, Bf, Cf) \
    __builtin_amdgcn_wmma_f32_16x16x32_bf16(false, (A), false, (Bf), (short)0, (Cf), false, false)

// ---------------------------------------------------------------------------
// fp32 -> bf16 streaming convert (4-wide)
// ---------------------------------------------------------------------------
__global__ void __launch_bounds__(256)
cvt_bf16_kernel(const float* __restrict__ src, __bf16* __restrict__ dst, int n4)
{
    int i = blockIdx.x * blockDim.x + threadIdx.x;
    if (i < n4) {
        f32x4 f = ((const f32x4*)src)[i];
        ((bf16x4*)dst)[i] = __builtin_convertvector(f, bf16x4);
    }
}

// q_learned[c] + pos_embed[p][c] -> bf16 [P][C]
__global__ void __launch_bounds__(256)
qpos_kernel(const float* __restrict__ ql, const float* __restrict__ pe,
            __bf16* __restrict__ dst, int n4)
{
    int i = blockIdx.x * blockDim.x + threadIdx.x;
    if (i < n4) {
        int c4 = i % (CC / 4);
        f32x4 q = ((const f32x4*)ql)[c4];
        f32x4 p = ((const f32x4*)pe)[i];
        ((bf16x4*)dst)[i] = __builtin_convertvector(q + p, bf16x4);
    }
}

// ---------------------------------------------------------------------------
// K/V projection GEMM: D[m][o] = sum_k A[m][k] * W[o][k]
// One wave computes a 32x64 tile (2x4 WMMA accumulators): 8 wmma / 12 b128.
// mode 0: store K as [B,H,N,HD]; mode 1: store V^T as [B,H,HD,N]
// ---------------------------------------------------------------------------
__global__ void __launch_bounds__(32)
gemm_kv_kernel(const __bf16* __restrict__ xbf, const __bf16* __restrict__ wbf,
               __bf16* __restrict__ out, int mode)
{
    const int lane = threadIdx.x & 31;
    const int lo = lane & 15, grp = lane >> 4;
    const int m0 = blockIdx.x * 32;
    const int o0 = blockIdx.y * 64;

    f32x8 acc[2][4] = {};
    const __bf16* arow[2];
    const __bf16* brow[4];
    arow[0] = xbf + (size_t)(m0 + lo) * CC;
    arow[1] = xbf + (size_t)(m0 + 16 + lo) * CC;
    for (int j = 0; j < 4; ++j) brow[j] = wbf + (size_t)(o0 + j * 16 + lo) * CC;

    for (int k0 = 0; k0 < CC; k0 += 32) {
        BF16x16U a[2], bf[4];
        for (int i = 0; i < 2; ++i) {
            a[i].h[0] = *(const bf16x8*)(arow[i] + k0 + grp * 8);
            a[i].h[1] = *(const bf16x8*)(arow[i] + k0 + grp * 8 + 16);
        }
        for (int j = 0; j < 4; ++j) {
            bf[j].h[0] = *(const bf16x8*)(brow[j] + k0 + grp * 8);
            bf[j].h[1] = *(const bf16x8*)(brow[j] + k0 + grp * 8 + 16);
        }
        for (int i = 0; i < 2; ++i)
            for (int j = 0; j < 4; ++j)
                acc[i][j] = WMMA_BF16(a[i].v, bf[j].v, acc[i][j]);
    }

    for (int i = 0; i < 2; ++i)
        for (int j = 0; j < 4; ++j) {
            int col = o0 + j * 16 + lo;
            int hh = col >> 6, d = col & 63;
            for (int r = 0; r < 8; ++r) {
                int m = m0 + i * 16 + r + 8 * grp;
                int bb = m >> 12, n = m & (NN - 1);
                size_t addr;
                if (mode == 0)
                    addr = (((size_t)(bb * HH + hh)) * NN + n) * HD + d;   // K  [B,H,N,HD]
                else
                    addr = (((size_t)(bb * HH + hh)) * HD + d) * NN + n;   // V^T[B,H,HD,N]
                out[addr] = (__bf16)acc[i][j][r];
            }
        }
}

// ---------------------------------------------------------------------------
// Fused flash attention.
// 8 waves per block share one (b,h); each wave owns 16 query rows.
// K/V^T chunks (32 n-columns) are staged into double-buffered LDS with
// global_load_async_to_lds_b128 (ASYNCcnt), then consumed via ds_load_b128
// WMMA fragments. Computes S^T = K*q^T so each softmax column p lives in
// lanes (p, p+16): reductions are a single shfl_xor(16).
// ---------------------------------------------------------------------------
__global__ void __launch_bounds__(256)
attn_kernel(const __bf16* __restrict__ qpos, const __bf16* __restrict__ kbuf,
            const __bf16* __restrict__ vtbuf, __bf16* __restrict__ attnout)
{
    __shared__ __bf16 kt[2][32 * 64];   // [n-local][d]      4 KB per buffer
    __shared__ __bf16 vt[2][64 * 32];   // [d][n-local]      4 KB per buffer

    const int tid  = threadIdx.x;
    const int lane = tid & 31;
    const int wave = tid >> 5;
    const int lo = lane & 15, grp = lane >> 4;
    const int h = blockIdx.y;
    const int b = blockIdx.z;
    const int p0 = blockIdx.x * (16 * ATT_WAVES) + wave * 16;

    const __bf16* Kb = kbuf  + ((size_t)(b * HH + h)) * NN * HD;
    const __bf16* Vt = vtbuf + ((size_t)(b * HH + h)) * HD * NN;

    // q^T B-frags: lane = column p, d at grp*8 and grp*8+16
    const __bf16* qrow = qpos + (size_t)(p0 + lo) * CC + h * HD + grp * 8;
    BF16x16U bq0, bq1;
    bq0.h[0] = *(const bf16x8*)(qrow);
    bq0.h[1] = *(const bf16x8*)(qrow + 16);
    bq1.h[0] = *(const bf16x8*)(qrow + 32);
    bq1.h[1] = *(const bf16x8*)(qrow + 48);

    // async stage of one 32-column chunk: 256 threads x 2 b128 = 8 KB
    auto stage = [&](int buf, int n0) {
        {
            unsigned lds = (unsigned)(unsigned long long)
                           &kt[buf][(tid >> 3) * 64 + (tid & 7) * 8];
            const void* g = (const void*)(Kb + (size_t)(n0 + (tid >> 3)) * HD + (tid & 7) * 8);
            asm volatile("global_load_async_to_lds_b128 %0, %1, off"
                         :: "v"(lds), "v"(g) : "memory");
        }
        {
            unsigned lds = (unsigned)(unsigned long long)
                           &vt[buf][(tid >> 2) * 32 + (tid & 3) * 8];
            const void* g = (const void*)(Vt + (size_t)(tid >> 2) * NN + n0 + (tid & 3) * 8);
            asm volatile("global_load_async_to_lds_b128 %0, %1, off"
                         :: "v"(lds), "v"(g) : "memory");
        }
    };

    stage(0, 0);

    float mrun = -1e30f, lrun = 0.f;
    f32x8 acc[4] = {};
    const int NCH = NN / 32;

    for (int i = 0; i < NCH; ++i) {
        const int buf = i & 1;
        if (i + 1 < NCH) {
            stage(buf ^ 1, (i + 1) * 32);
            asm volatile("s_wait_asynccnt 0x2" ::: "memory"); // this wave's chunk i resident
        } else {
            asm volatile("s_wait_asynccnt 0x0" ::: "memory");
        }
        __syncthreads();  // all waves' pieces of chunk i resident

        const __bf16* kk = &kt[buf][0];
        const __bf16* vv = &vt[buf][0];

        // ---- S^T tiles (rows n-local, cols p) from LDS ----
        BF16x16U a;
        f32x8 s0 = {}, s1 = {};
        a.h[0] = *(const bf16x8*)(kk + lo * 64 + grp * 8);
        a.h[1] = *(const bf16x8*)(kk + lo * 64 + grp * 8 + 16);
        s0 = WMMA_BF16(a.v, bq0.v, s0);
        a.h[0] = *(const bf16x8*)(kk + lo * 64 + 32 + grp * 8);
        a.h[1] = *(const bf16x8*)(kk + lo * 64 + 32 + grp * 8 + 16);
        s0 = WMMA_BF16(a.v, bq1.v, s0);
        a.h[0] = *(const bf16x8*)(kk + (16 + lo) * 64 + grp * 8);
        a.h[1] = *(const bf16x8*)(kk + (16 + lo) * 64 + grp * 8 + 16);
        s1 = WMMA_BF16(a.v, bq0.v, s1);
        a.h[0] = *(const bf16x8*)(kk + (16 + lo) * 64 + 32 + grp * 8);
        a.h[1] = *(const bf16x8*)(kk + (16 + lo) * 64 + 32 + grp * 8 + 16);
        s1 = WMMA_BF16(a.v, bq1.v, s1);

        // ---- online softmax over this 32-row n chunk (per column p) ----
        float cm = -1e30f;
        float e0[8], e1[8];
        for (int r = 0; r < 8; ++r) {
            s0[r] *= QK_SCALE; s1[r] *= QK_SCALE;
            cm = fmaxf(cm, fmaxf(s0[r], s1[r]));
        }
        cm = fmaxf(cm, __shfl_xor(cm, 16, 32));
        float mnew  = fmaxf(mrun, cm);
        float alpha = __expf(mrun - mnew);
        mrun = mnew;
        float ssum = 0.f;
        for (int r = 0; r < 8; ++r) {
            e0[r] = __expf(s0[r] - mnew);
            e1[r] = __expf(s1[r] - mnew);
            ssum += e0[r] + e1[r];
        }
        lrun = lrun * alpha + ssum + __shfl_xor(ssum, 16, 32);

        // ---- repack exp(S^T) into B-frag of P^T via lane-pair exchange ----
        BF16x16U pb;
        for (int r = 0; r < 8; ++r) {
            float oth0 = __shfl_xor(e0[r], 16, 32);
            float oth1 = __shfl_xor(e1[r], 16, 32);
            float first  = grp ? oth1 : e0[r];
            float second = grp ? e1[r] : oth0;
            pb.v[r]     = (__bf16)first;
            pb.v[r + 8] = (__bf16)second;
        }

        // ---- out^T += V^T(d-tile, n-chunk) * P^T, V^T from LDS ----
        for (int t = 0; t < 4; ++t) {
            for (int r = 0; r < 8; ++r) acc[t][r] *= alpha;
            BF16x16U av;
            av.h[0] = *(const bf16x8*)(vv + (t * 16 + lo) * 32 + grp * 16);
            av.h[1] = *(const bf16x8*)(vv + (t * 16 + lo) * 32 + grp * 16 + 8);
            acc[t] = WMMA_BF16(av.v, pb.v, acc[t]);
        }

        __syncthreads();  // done reading buf before it is overwritten
    }

    float inv = 1.0f / lrun;
    __bf16* orow = attnout + ((size_t)(b * PP) + p0 + lo) * CC + h * HD;
    for (int t = 0; t < 4; ++t)
        for (int r = 0; r < 8; ++r) {
            int d = t * 16 + r + 8 * grp;
            orow[d] = (__bf16)(acc[t][r] * inv);
        }
}

// ---------------------------------------------------------------------------
// Final projection: out[m][o] = sum_k A[m][k]*W[o][k] + bias[o], fp32 output
// 32x64 tile per wave.
// ---------------------------------------------------------------------------
__global__ void __launch_bounds__(32)
gemm_proj_kernel(const __bf16* __restrict__ abf, const __bf16* __restrict__ wbf,
                 const float* __restrict__ bias, float* __restrict__ out)
{
    const int lane = threadIdx.x & 31;
    const int lo = lane & 15, grp = lane >> 4;
    const int m0 = blockIdx.x * 32;
    const int o0 = blockIdx.y * 64;

    f32x8 acc[2][4] = {};
    const __bf16* arow[2];
    const __bf16* brow[4];
    arow[0] = abf + (size_t)(m0 + lo) * CC;
    arow[1] = abf + (size_t)(m0 + 16 + lo) * CC;
    for (int j = 0; j < 4; ++j) brow[j] = wbf + (size_t)(o0 + j * 16 + lo) * CC;

    for (int k0 = 0; k0 < CC; k0 += 32) {
        BF16x16U a[2], bf[4];
        for (int i = 0; i < 2; ++i) {
            a[i].h[0] = *(const bf16x8*)(arow[i] + k0 + grp * 8);
            a[i].h[1] = *(const bf16x8*)(arow[i] + k0 + grp * 8 + 16);
        }
        for (int j = 0; j < 4; ++j) {
            bf[j].h[0] = *(const bf16x8*)(brow[j] + k0 + grp * 8);
            bf[j].h[1] = *(const bf16x8*)(brow[j] + k0 + grp * 8 + 16);
        }
        for (int i = 0; i < 2; ++i)
            for (int j = 0; j < 4; ++j)
                acc[i][j] = WMMA_BF16(a[i].v, bf[j].v, acc[i][j]);
    }

    for (int i = 0; i < 2; ++i)
        for (int j = 0; j < 4; ++j) {
            int col = o0 + j * 16 + lo;
            float bv = bias[col];
            for (int r = 0; r < 8; ++r) {
                int m = m0 + i * 16 + r + 8 * grp;
                out[(size_t)m * CC + col] = acc[i][j][r] + bv;
            }
        }
}

// ---------------------------------------------------------------------------
extern "C" void kernel_launch(void* const* d_in, const int* in_sizes, int n_in,
                              void* d_out, int out_size, void* d_ws, size_t ws_size,
                              hipStream_t stream)
{
    const float* x  = (const float*)d_in[0];   // [B,N,C]
    const float* wk = (const float*)d_in[1];   // [C,C]
    const float* wv = (const float*)d_in[2];   // [C,C]
    const float* ql = (const float*)d_in[3];   // [1,1,C]
    const float* pe = (const float*)d_in[4];   // [1,P,C]
    const float* wp = (const float*)d_in[5];   // [C,C]
    const float* pb = (const float*)d_in[6];   // [C]
    float* out = (float*)d_out;                // [B,P,C] fp32

    char* ws = (char*)d_ws;
    size_t off = 0;
    auto alloc = [&](size_t bytes) -> char* {
        char* p = ws + off;
        off += (bytes + 255) & ~(size_t)255;
        return p;
    };
    __bf16* xbf   = (__bf16*)alloc((size_t)BB * NN * CC * 2);        // 24 MB
    __bf16* wkbf  = (__bf16*)alloc((size_t)CC * CC * 2);
    __bf16* wvbf  = (__bf16*)alloc((size_t)CC * CC * 2);
    __bf16* wpbf  = (__bf16*)alloc((size_t)CC * CC * 2);
    __bf16* qposb = (__bf16*)alloc((size_t)PP * CC * 2);
    __bf16* kbuf  = (__bf16*)alloc((size_t)BB * HH * NN * HD * 2);   // 24 MB
    __bf16* vtbuf = (__bf16*)alloc((size_t)BB * HH * HD * NN * 2);   // 24 MB
    __bf16* attnb = (__bf16*)alloc((size_t)BB * PP * CC * 2);        // 6 MB

    // bf16 conversions
    {
        int n4 = BB * NN * CC / 4;
        cvt_bf16_kernel<<<(n4 + 255) / 256, 256, 0, stream>>>(x, xbf, n4);
        int w4 = CC * CC / 4;
        cvt_bf16_kernel<<<(w4 + 255) / 256, 256, 0, stream>>>(wk, wkbf, w4);
        cvt_bf16_kernel<<<(w4 + 255) / 256, 256, 0, stream>>>(wv, wvbf, w4);
        cvt_bf16_kernel<<<(w4 + 255) / 256, 256, 0, stream>>>(wp, wpbf, w4);
        int q4 = PP * CC / 4;
        qpos_kernel<<<(q4 + 255) / 256, 256, 0, stream>>>(ql, pe, qposb, q4);
    }

    // K and V^T projections (WMMA bf16)
    {
        dim3 grid(BB * NN / 32, CC / 64);
        gemm_kv_kernel<<<grid, 32, 0, stream>>>(xbf, wkbf, kbuf, 0);
        gemm_kv_kernel<<<grid, 32, 0, stream>>>(xbf, wvbf, vtbuf, 1);
    }

    // fused flash attention (async LDS staging, 8 waves share one (b,h))
    {
        dim3 grid(PP / (16 * ATT_WAVES), HH, BB);
        attn_kernel<<<grid, 256, 0, stream>>>(qposb, kbuf, vtbuf, attnb);
    }

    // final projection + bias
    {
        dim3 grid(BB * PP / 32, CC / 64);
        gemm_proj_kernel<<<grid, 32, 0, stream>>>(attnb, wpbf, pb, out);
    }
}
